// AttentionDecoderRNN_89077621719299
// MI455X (gfx1250) — compile-verified, matrix operand order checked by reading, not run on hardware
//
#include <hip/hip_runtime.h>

// Problem constants (match reference).
#define T_STEPS 200
#define BB 128
#define SS 512
#define HH 512
#define VV 2048

typedef __attribute__((ext_vector_type(16))) _Float16 v16h;
typedef __attribute__((ext_vector_type(8)))  _Float16 v8h;
typedef __attribute__((ext_vector_type(8)))  float    v8f;

union FragH { v16h v; struct { v8h lo, hi; } p; };

// ---------------------------------------------------------------------------
// WMMA helpers (gfx1250: V_WMMA_F32_16X16X32_F16, fp32 accumulate)
// ---------------------------------------------------------------------------
__device__ __forceinline__ v8f wmma_f16(v16h a, v16h b, v8f c) {
  return __builtin_amdgcn_wmma_f32_16x16x32_f16(
      /*neg_a=*/false, a, /*neg_b=*/false, b,
      /*c_mod=*/(short)0, c, /*reuse_a=*/false, /*reuse_b=*/false);
}

// B fragment: row-major f16 weight row (col n owned by this lane), two aligned
// 16B loads at K offsets {8*half, 16+8*half}.
__device__ __forceinline__ v16h loadB_f16(const _Float16* __restrict__ row, int half) {
  FragH f;
  f.p.lo = *(const v8h*)(row + 8 * half);
  f.p.hi = *(const v8h*)(row + 16 + 8 * half);
  return f.v;
}

// A fragment from fp32 activations, converted to f16 in-register. Source row
// is split into two segments: k < KA reads from ra, k >= KA reads from rb
// (used for the [emb | ctx] and [h2 | ctx] concatenations without
// materializing them). Run length is 8 and KA is a multiple of 8, so a run
// never straddles the boundary.
__device__ __forceinline__ v16h loadA_f32(const float* __restrict__ ra,
                                          const float* __restrict__ rb,
                                          int KA, int kb, int half) {
  const int k0 = kb + 8 * half;
  const int k1 = kb + 16 + 8 * half;
  const float* p0 = (k0 < KA) ? (ra + k0) : (rb + (k0 - KA));
  const float* p1 = (k1 < KA) ? (ra + k1) : (rb + (k1 - KA));
  v8f r0 = *(const v8f*)p0;
  v8f r1 = *(const v8f*)p1;
  v16h a;
#pragma unroll
  for (int i = 0; i < 8; ++i) {
    a[i]     = (_Float16)r0[i];
    a[8 + i] = (_Float16)r1[i];
  }
  return a;
}

// ---------------------------------------------------------------------------
// Fused GRU layer: hnew = GRUCell([xA|xB], hprev). One wave per 16x16 h-tile,
// six fp32 WMMA accumulators (r/z/n gates for Wi and Wh paths), gate math in
// the epilogue on accumulator registers. Optionally emits tanh(hnew).
// Grid: (B/16, H/64), block 128 (4 waves).
// ---------------------------------------------------------------------------
__global__ void __launch_bounds__(128)
k_gru(const float* __restrict__ embTab, const int* __restrict__ yt,
      const float* __restrict__ xA, int KA,
      const float* __restrict__ xB, int KB,
      const float* __restrict__ hprev,
      const _Float16* __restrict__ Wi, const _Float16* __restrict__ Wh,
      const float* __restrict__ bi, const float* __restrict__ bh,
      float* __restrict__ hnew, float* __restrict__ th_out) {
  const int lane = threadIdx.x & 31;
  const int wv   = threadIdx.x >> 5;
  const int half = lane >> 4;
  const int l16  = lane & 15;
  const int bt = blockIdx.x;
  const int jt = blockIdx.y * 4 + wv;
  const int Kx = KA + KB;
  const int bA = bt * 16 + l16;   // row this lane feeds into A fragments
  const int j  = jt * 16 + l16;   // output column this lane owns

  const float* ra = yt ? (embTab + (size_t)yt[bA] * KA)
                       : (xA + (size_t)bA * KA);
  const float* rb = xB ? (xB + (size_t)bA * KB) : nullptr;
  const float* rh = hprev + (size_t)bA * HH;

  const _Float16* wi_r = Wi + (size_t)j * Kx;
  const _Float16* wi_z = Wi + (size_t)(j + HH) * Kx;
  const _Float16* wi_n = Wi + (size_t)(j + 2 * HH) * Kx;
  const _Float16* wh_r = Wh + (size_t)j * HH;
  const _Float16* wh_z = Wh + (size_t)(j + HH) * HH;
  const _Float16* wh_n = Wh + (size_t)(j + 2 * HH) * HH;

  v8f aR = {}, aZ = {}, aN = {}, gR = {}, gZ = {}, gN = {};
  for (int kb = 0; kb < Kx; kb += 32) {
    v16h a = loadA_f32(ra, rb, KA, kb, half);
    aR = wmma_f16(a, loadB_f16(wi_r + kb, half), aR);
    aZ = wmma_f16(a, loadB_f16(wi_z + kb, half), aZ);
    aN = wmma_f16(a, loadB_f16(wi_n + kb, half), aN);
  }
  for (int kb = 0; kb < HH; kb += 32) {
    v16h a = loadA_f32(rh, nullptr, HH, kb, half);
    gR = wmma_f16(a, loadB_f16(wh_r + kb, half), gR);
    gZ = wmma_f16(a, loadB_f16(wh_z + kb, half), gZ);
    gN = wmma_f16(a, loadB_f16(wh_n + kb, half), gN);
  }

  const float biR = bi[j], biZ = bi[j + HH], biN = bi[j + 2 * HH];
  const float bhR = bh[j], bhZ = bh[j + HH], bhN = bh[j + 2 * HH];
#pragma unroll
  for (int r = 0; r < 8; ++r) {
    const int b = bt * 16 + r + 8 * half;   // C/D layout: M = r + 8*(lane>=16)
    float ir = aR[r] + biR, iz = aZ[r] + biZ, in = aN[r] + biN;
    float hr = gR[r] + bhR, hz = gZ[r] + bhZ, hn = gN[r] + bhN;
    float rg = 1.0f / (1.0f + expf(-(ir + hr)));
    float zg = 1.0f / (1.0f + expf(-(iz + hz)));
    float ng = tanhf(in + rg * hn);
    float hp = hprev[(size_t)b * HH + j];
    float h  = (1.0f - zg) * ng + zg * hp;
    hnew[(size_t)b * HH + j] = h;
    if (th_out) th_out[(size_t)b * HH + j] = tanhf(h);
  }
}

// ---------------------------------------------------------------------------
// Output projection: out[b,v] = [h2|ctx] . Wc[v,:] + bc[v].
// Grid: (B/16, V/64), block 128.
// ---------------------------------------------------------------------------
__global__ void __launch_bounds__(128)
k_outproj(const float* __restrict__ h2, const float* __restrict__ ctx,
          const _Float16* __restrict__ Wc, const float* __restrict__ bc,
          float* __restrict__ out) {
  const int lane = threadIdx.x & 31;
  const int wv   = threadIdx.x >> 5;
  const int half = lane >> 4;
  const int l16  = lane & 15;
  const int bt = blockIdx.x;
  const int vt = blockIdx.y * 4 + wv;
  const int bA = bt * 16 + l16;
  const int v  = vt * 16 + l16;

  const float* ra = h2 + (size_t)bA * HH;
  const float* rb = ctx + (size_t)bA * HH;
  const _Float16* wr = Wc + (size_t)v * (2 * HH);

  v8f acc = {};
  for (int kb = 0; kb < 2 * HH; kb += 32) {
    v16h a = loadA_f32(ra, rb, HH, kb, half);
    acc = wmma_f16(a, loadB_f16(wr + kb, half), acc);
  }
  const float bv = bc[v];
#pragma unroll
  for (int r = 0; r < 8; ++r) {
    const int b = bt * 16 + r + 8 * half;
    out[(size_t)b * VV + v] = acc[r] + bv;
  }
}

// ---------------------------------------------------------------------------
// Fused attention: scores -> softmax -> context, one block per b.
//   phase 1 (thread = s): scores[b,s] = sum_h th2[b,h]*tanh_enc[s,b,h]
//                         softmax over s, attw kept in LDS + written to out
//   phase 2 (thread = h): ctx[b,h]    = sum_s attw[s]*enc[s,b,h]
// Not GEMM-shaped (enc is indexed by b), so VALU with LDS staging. Fusing
// removes one launch per step and a [B,S] global round-trip.
// ---------------------------------------------------------------------------
__global__ void __launch_bounds__(512)
k_attn_ctx(const float* __restrict__ th2, const _Float16* __restrict__ tenc,
           const _Float16* __restrict__ yench, float* __restrict__ attw,
           float* __restrict__ ctxn, int t) {
  __shared__ float sh[HH];    // th2 row
  __shared__ float red[SS];   // reduction scratch
  __shared__ float aw[SS];    // attention weights
  const int b = blockIdx.x;
  const int s = threadIdx.x;  // == h in phase 2 (blockDim = 512 = S = H)

  sh[s] = th2[(size_t)b * HH + s];
  __syncthreads();

  const _Float16* e = tenc + ((size_t)s * BB + b) * HH;
  float acc = 0.f;
  for (int h = 0; h < HH; h += 8) {
    v8h x = *(const v8h*)(e + h);
#pragma unroll
    for (int i = 0; i < 8; ++i) acc += sh[h + i] * (float)x[i];
  }

  red[s] = acc; __syncthreads();
  for (int off = 256; off > 0; off >>= 1) {
    if (s < off) red[s] = fmaxf(red[s], red[s + off]);
    __syncthreads();
  }
  const float mx = red[0]; __syncthreads();
  const float ex = expf(acc - mx);
  red[s] = ex; __syncthreads();
  for (int off = 256; off > 0; off >>= 1) {
    if (s < off) red[s] += red[s + off];
    __syncthreads();
  }
  const float w = ex * (1.0f / red[0]);
  aw[s] = w;
  attw[(size_t)b * T_STEPS * SS + (size_t)t * SS + s] = w;
  __syncthreads();

  // phase 2: context reduction over s (thread index now plays the role of h)
  const int h = s;
  float c = 0.f;
  for (int si = 0; si < SS; ++si)
    c += aw[si] * (float)yench[((size_t)si * BB + b) * HH + h];
  ctxn[(size_t)b * HH + h] = c;
}

// ---------------------------------------------------------------------------
// Cross-entropy: per-row NLL (deterministic), then single-block reduce.
// ---------------------------------------------------------------------------
__global__ void __launch_bounds__(256)
k_rownll(const float* __restrict__ logits, const int* __restrict__ labels,
         float* __restrict__ nll, float* __restrict__ msk) {
  __shared__ float red[256];
  const size_t row = blockIdx.x;
  const float* p = logits + row * VV;
  const int tid = threadIdx.x;
  float mx = -3.402823466e38f;
  for (int i = tid; i < VV; i += 256) mx = fmaxf(mx, p[i]);
  red[tid] = mx; __syncthreads();
  for (int off = 128; off > 0; off >>= 1) {
    if (tid < off) red[tid] = fmaxf(red[tid], red[tid + off]);
    __syncthreads();
  }
  mx = red[0]; __syncthreads();
  float sm = 0.f;
  for (int i = tid; i < VV; i += 256) sm += expf(p[i] - mx);
  red[tid] = sm; __syncthreads();
  for (int off = 128; off > 0; off >>= 1) {
    if (tid < off) red[tid] += red[tid + off];
    __syncthreads();
  }
  if (tid == 0) {
    const int lab = labels[row];
    const float lse = mx + logf(red[0]);
    const float m = (lab != 0) ? 1.0f : 0.0f;   // PAD_TOKEN == 0
    nll[row] = (lse - p[lab]) * m;
    msk[row] = m;
  }
}

__global__ void __launch_bounds__(256)
k_ce(const float* __restrict__ nll, const float* __restrict__ msk, int n,
     float* __restrict__ out) {
  __shared__ float r0[256], r1[256];
  const int tid = threadIdx.x;
  float s = 0.f, c = 0.f;
  for (int i = tid; i < n; i += 256) { s += nll[i]; c += msk[i]; }
  r0[tid] = s; r1[tid] = c; __syncthreads();
  for (int off = 128; off > 0; off >>= 1) {
    if (tid < off) { r0[tid] += r0[tid + off]; r1[tid] += r1[tid + off]; }
    __syncthreads();
  }
  if (tid == 0) out[0] = r0[0] / fmaxf(r1[0], 1.0f);
}

// ---------------------------------------------------------------------------
// One-time prep: f32->f16 weight conversion, enc -> (enc_f16, tanh(enc)_f16),
// and zeroing of hidden/ctx state.
// ---------------------------------------------------------------------------
__global__ void k_cvt(const float* __restrict__ s, _Float16* __restrict__ d,
                      long long n) {
  long long i = (long long)blockIdx.x * blockDim.x + threadIdx.x;
  const long long st = (long long)gridDim.x * blockDim.x;
  for (; i < n; i += st) d[i] = (_Float16)s[i];
}

__global__ void k_prep_enc(const float* __restrict__ s,
                           _Float16* __restrict__ y, _Float16* __restrict__ tn,
                           long long n) {
  long long i = (long long)blockIdx.x * blockDim.x + threadIdx.x;
  const long long st = (long long)gridDim.x * blockDim.x;
  for (; i < n; i += st) {
    const float v = s[i];
    y[i]  = (_Float16)v;
    tn[i] = (_Float16)tanhf(v);
  }
}

__global__ void k_zero(float* __restrict__ p, long long n) {
  long long i = (long long)blockIdx.x * blockDim.x + threadIdx.x;
  const long long st = (long long)gridDim.x * blockDim.x;
  for (; i < n; i += st) p[i] = 0.f;
}

// ---------------------------------------------------------------------------
// Host orchestration: ~1000 graph nodes (5 kernels x 200 steps + prep).
// Workspace: fp32 state (h ping-pong x3, ctx ping-pong, th2, CE rows) then
// f16 weights + f16 enc copies (~152 MB). Keeping enc as f16 (2 x 67 MB)
// plus ~15 MB of weights keeps the per-step 134 MB encoder traffic resident
// in the 192 MB L2 instead of hitting HBM every step.
// ---------------------------------------------------------------------------
extern "C" void kernel_launch(void* const* d_in, const int* in_sizes, int n_in,
                              void* d_out, int out_size, void* d_ws,
                              size_t ws_size, hipStream_t stream) {
  (void)in_sizes; (void)n_in; (void)out_size; (void)ws_size;

  const int*   y_in  = (const int*)d_in[0];
  const int*   y_lab = (const int*)d_in[1];
  const float* y_enc = (const float*)d_in[2];
  const float* emb   = (const float*)d_in[3];
  const float* Wi0 = (const float*)d_in[4];  const float* Wh0 = (const float*)d_in[5];
  const float* bi0 = (const float*)d_in[6];  const float* bh0 = (const float*)d_in[7];
  const float* Wi1 = (const float*)d_in[8];  const float* Wh1 = (const float*)d_in[9];
  const float* bi1 = (const float*)d_in[10]; const float* bh1 = (const float*)d_in[11];
  const float* Wi2 = (const float*)d_in[12]; const float* Wh2 = (const float*)d_in[13];
  const float* bi2 = (const float*)d_in[14]; const float* bh2 = (const float*)d_in[15];
  const float* Wc  = (const float*)d_in[16]; const float* bc  = (const float*)d_in[17];

  float* out  = (float*)d_out;
  float* pred = out;                                          // [T,B,V]
  float* att  = out + (size_t)T_STEPS * BB * VV;              // [B,T,S]
  float* ce   = att + (size_t)BB * T_STEPS * SS;              // [1]

  const size_t N = (size_t)BB * HH;                           // 65536
  float* f = (float*)d_ws;
  float* hbuf = f;                        // 8*N: h0a h0b h1a h1b h2a h2b ctxA ctxB
  float* th2  = f + 8 * N;                // N
  float* nllb = f + 9 * N;                // T*B
  float* mskb = nllb + (size_t)T_STEPS * BB;
  _Float16* hb = (_Float16*)(f + 9 * N + 2 * (size_t)T_STEPS * BB);

  _Float16* Wi0h = hb;                      // 1536*1024
  _Float16* Wh0h = Wi0h + (size_t)1536 * 1024;
  _Float16* Wi1h = Wh0h + (size_t)1536 * 512;
  _Float16* Wh1h = Wi1h + (size_t)1536 * 512;
  _Float16* Wi2h = Wh1h + (size_t)1536 * 512;
  _Float16* Wh2h = Wi2h + (size_t)1536 * 512;
  _Float16* Wch  = Wh2h + (size_t)1536 * 512;   // 2048*1024
  _Float16* ych  = Wch + (size_t)2048 * 1024;   // S*B*H
  _Float16* tch  = ych + (size_t)SS * BB * HH;  // S*B*H

  // --- one-time prep ---
  k_cvt<<<1024, 256, 0, stream>>>(Wi0, Wi0h, (long long)1536 * 1024);
  k_cvt<<<1024, 256, 0, stream>>>(Wh0, Wh0h, (long long)1536 * 512);
  k_cvt<<<1024, 256, 0, stream>>>(Wi1, Wi1h, (long long)1536 * 512);
  k_cvt<<<1024, 256, 0, stream>>>(Wh1, Wh1h, (long long)1536 * 512);
  k_cvt<<<1024, 256, 0, stream>>>(Wi2, Wi2h, (long long)1536 * 512);
  k_cvt<<<1024, 256, 0, stream>>>(Wh2, Wh2h, (long long)1536 * 512);
  k_cvt<<<1024, 256, 0, stream>>>(Wc, Wch, (long long)2048 * 1024);
  k_prep_enc<<<4096, 256, 0, stream>>>(y_enc, ych, tch,
                                       (long long)SS * BB * HH);
  k_zero<<<2048, 256, 0, stream>>>(hbuf, (long long)8 * N);

  float* hprev[3] = {hbuf + 0 * N, hbuf + 2 * N, hbuf + 4 * N};
  float* hnext[3] = {hbuf + 1 * N, hbuf + 3 * N, hbuf + 5 * N};
  float* ctx_cur = hbuf + 6 * N;
  float* ctx_nxt = hbuf + 7 * N;

  const dim3 gGRU(BB / 16, HH / 64);     // 8 x 8
  const dim3 gOUT(BB / 16, VV / 64);     // 8 x 32

  for (int t = 0; t < T_STEPS; ++t) {
    // layer 0: x = [emb[y_t] | ctx]
    k_gru<<<gGRU, 128, 0, stream>>>(emb, y_in + (size_t)t * BB,
                                    nullptr, HH, ctx_cur, HH,
                                    hprev[0], Wi0h, Wh0h, bi0, bh0,
                                    hnext[0], nullptr);
    // layer 1
    k_gru<<<gGRU, 128, 0, stream>>>(nullptr, nullptr, hnext[0], HH,
                                    nullptr, 0, hprev[1], Wi1h, Wh1h,
                                    bi1, bh1, hnext[1], nullptr);
    // layer 2 (+ tanh(h2) for attention)
    k_gru<<<gGRU, 128, 0, stream>>>(nullptr, nullptr, hnext[1], HH,
                                    nullptr, 0, hprev[2], Wi2h, Wh2h,
                                    bi2, bh2, hnext[2], th2);
    // fused attention weights + context
    k_attn_ctx<<<BB, SS, 0, stream>>>(th2, tch, ych, att, ctx_nxt, t);
    // logits for this step
    k_outproj<<<gOUT, 128, 0, stream>>>(hnext[2], ctx_nxt, Wch, bc,
                                        pred + (size_t)t * BB * VV);
    // ping-pong
    for (int l = 0; l < 3; ++l) { float* tmp = hprev[l]; hprev[l] = hnext[l]; hnext[l] = tmp; }
    { float* tmp = ctx_cur; ctx_cur = ctx_nxt; ctx_nxt = tmp; }
  }

  // cross-entropy (deterministic two-pass reduction)
  k_rownll<<<T_STEPS * BB, 256, 0, stream>>>(pred, y_lab, nllb, mskb);
  k_ce<<<1, 256, 0, stream>>>(nllb, mskb, T_STEPS * BB, ce);
}
